// Attention_9088150798968
// MI455X (gfx1250) — matrix operand order repaired and checked by reading, not compiled, not run on hardware
//
#include <hip/hip_runtime.h>
#include <hip/hip_bf16.h>

// ---------------------------------------------------------------------------
// Problem constants (match reference)
// ---------------------------------------------------------------------------
#define BB 2
#define TT 2048
#define CC 1024
#define HH 16
#define DD 64
#define MM (BB * TT)      // 4096 rows for the token-major GEMMs
#define N_QKV (3 * CC)    // 3072

#define LDS_STRIDE 40     // bf16 elems per LDS slab row (80B: pads 64B rows)

typedef __bf16 v16bf __attribute__((ext_vector_type(16)));
typedef float  v8f   __attribute__((ext_vector_type(8)));
typedef unsigned int v4u __attribute__((ext_vector_type(4)));
typedef unsigned int v8u __attribute__((ext_vector_type(8)));

// ---------------------------------------------------------------------------
// WMMA helper: D = A(16x32 bf16) * B(32x16 bf16) + C(16x16 f32)
// ---------------------------------------------------------------------------
__device__ __forceinline__ v8f wmma_bf16(v16bf a, v16bf b, v8f c) {
  return __builtin_amdgcn_wmma_f32_16x16x32_bf16(
      /*neg_a=*/false, a, /*neg_b=*/false, b,
      /*c_mod=*/(short)0, c, /*reuse_a=*/false, /*reuse_b=*/false);
}

// A-matrix (16x32, row-major MxK source):
//   lanes 0-15 : row M = lane,    halves 0-15 -> K k0+0..15 (contiguous)
//   lanes 16-31: row M = lane-16, halves 0-15 -> K k0+16..31 (contiguous)
__device__ __forceinline__ v16bf load_a16(const __bf16* base, int ld, int r0, int k0) {
  int lane = threadIdx.x & 31;
  const __bf16* p = base + (size_t)(r0 + (lane & 15)) * ld + k0 + ((lane >> 4) << 4);
  uint4 lo = *reinterpret_cast<const uint4*>(p);
  uint4 hi = *reinterpret_cast<const uint4*>(p + 8);
  v16bf out;
  reinterpret_cast<uint4*>(&out)[0] = lo;
  reinterpret_cast<uint4*>(&out)[1] = hi;
  return out;
}

// B-matrix (32x16) loaded from the N-major (transposed) source Bt[n][k]:
//   lanes 0-15 : col N = lane,    halves 0-15 -> K k0+0..15 (contiguous)
//   lanes 16-31: col N = lane-16, halves 0-15 -> K k0+16..31 (contiguous)
__device__ __forceinline__ v16bf load_b16(const __bf16* base, int ld, int n0, int k0) {
  int lane = threadIdx.x & 31;
  const __bf16* p = base + (size_t)(n0 + (lane & 15)) * ld + k0 + ((lane >> 4) << 4);
  uint4 lo = *reinterpret_cast<const uint4*>(p);
  uint4 hi = *reinterpret_cast<const uint4*>(p + 8);
  v16bf out;
  reinterpret_cast<uint4*>(&out)[0] = lo;
  reinterpret_cast<uint4*>(&out)[1] = hi;
  return out;
}

// gfx1250 async LDS fill: GLOBAL_LOAD_ASYNC_TO_LDS_B128 (ASYNCcnt-tracked).
// VDST = LDS byte address (addr[31:0] of the generic pointer), VADDR = 64-bit
// global address, GV mode (saddr = off).
__device__ __forceinline__ void async_copy_b128(uint32_t lds_addr, const void* gptr) {
  asm volatile("global_load_async_to_lds_b128 %0, %1, off"
               :
               : "v"(lds_addr), "v"(gptr)
               : "memory");
}

__device__ __forceinline__ void wait_async_all() {
  asm volatile("s_wait_asynccnt 0" ::: "memory");
}

// gfx1250 Tensor Data Mover: one descriptor pulls a rows x row_elems bf16 tile
// from a row-major tensor (leading dim ld_elems) into LDS, inserting 16B of
// LDS padding after every 64B so rows land on an 80B (LDS_STRIDE) pitch.
// D# fields per CDNA5 ISA; all inputs are
// workgroup-uniform so the descriptor groups live in SGPRs as required.
__device__ __forceinline__ void tdm_load_slab_bf16(
    uint32_t lds_addr, const void* gptr,
    uint32_t row_elems, uint32_t rows, uint32_t ld_elems,
    uint32_t tensor_d0, uint32_t tensor_d1) {
  uint64_t ga = (uint64_t)(uintptr_t)gptr;
  v4u g0;
  g0[0] = 1u;                                   // count=1, user descriptor
  g0[1] = lds_addr;                             // lds_addr -> bits 63:32
  g0[2] = (uint32_t)ga;                         // global_addr[31:0]
  g0[3] = (uint32_t)((ga >> 32) & 0x1FFFFFFu)   // global_addr[56:32]
        | (2u << 30);                           // type=2 ("image")
  v8u g1;
  g1[0] = (1u << 16)     // data_size = 1 -> 2 bytes
        | (1u << 20)     // pad_enable
        | (3u << 22)     // pad_interval = 16 DWORDs (64B) between pads
        | (3u << 25);    // pad_amount   = 4 DWORDs (16B)
  g1[1] = (tensor_d0 & 0xFFFFu) << 16;          // tensor_dim0[15:0] -> bits 63:48
  g1[2] = (tensor_d0 >> 16)                     // tensor_dim0[31:16]
        | ((tensor_d1 & 0xFFFFu) << 16);        // tensor_dim1[15:0]
  g1[3] = (tensor_d1 >> 16)                     // tensor_dim1[31:16]
        | (row_elems << 16);                    // tile_dim0 -> bits 127:112
  g1[4] = rows;                                 // tile_dim1; tile_dim2=0
  g1[5] = ld_elems;                             // tensor_dim0_stride[31:0]
  g1[6] = 0u;                                   // stride[47:32]=0, dim1_stride lo=0
  g1[7] = 0u;                                   // dim1_stride hi = 0
  asm volatile("tensor_load_to_lds %0, %1" :: "s"(g0), "s"(g1) : "memory");
}

// ---------------------------------------------------------------------------
// f32 -> bf16 conversion
// ---------------------------------------------------------------------------
__global__ void f32_to_bf16_kernel(const float* __restrict__ in,
                                   __bf16* __restrict__ out, int n) {
  int i = blockIdx.x * blockDim.x + threadIdx.x;
  if (i < n) out[i] = (__bf16)in[i];
}

// ---------------------------------------------------------------------------
// LDS-staged GEMM: C[m][n] = sum_k A[m][k] * Bt[n][k] + bias[n]
// 256 threads / 8 waves per block; 128x128 output tile per block.
// Wave (wm, wn) in a 2x4 grid owns a 64x32 register tile = 4x2 WMMA tiles.
// Per K-step (BK=32): A slab is fetched by the Tensor Data Mover (one D#
// issued by wave 0, TENSORcnt-tracked), B slab by per-lane async-to-LDS
// copies (ASYNCcnt-tracked) -- the two fills ride different engines.
// ---------------------------------------------------------------------------
template <typename OutT>
__global__ __launch_bounds__(256) void wmma_gemm_async(
    const __bf16* __restrict__ A, const __bf16* __restrict__ Bt,
    const float* __restrict__ bias, OutT* __restrict__ C,
    int M, int N, int K) {
  __shared__ __attribute__((aligned(16))) __bf16 As[128 * LDS_STRIDE];
  __shared__ __attribute__((aligned(16))) __bf16 Bs[128 * LDS_STRIDE];

  int tid = threadIdx.x;
  int wave = tid >> 5, lane = tid & 31;
  int wm = wave >> 2, wn = wave & 3;
  int m0 = blockIdx.y * 128, n0 = blockIdx.x * 128;

  // B slab: 128x32 bf16 = 512 x 16B chunks; 256 threads copy 2 chunks each.
  int c0 = tid, c1 = tid + 256;
  int r0 = c0 >> 2, x0 = c0 & 3;
  int r1 = c1 >> 2, x1 = c1 & 3;

  const __bf16* gb0 = Bt + (size_t)(n0 + r0) * K + x0 * 8;
  const __bf16* gb1 = Bt + (size_t)(n0 + r1) * K + x1 * 8;
  uint32_t lb0 = (uint32_t)(uintptr_t)&Bs[r0 * LDS_STRIDE + x0 * 8];
  uint32_t lb1 = (uint32_t)(uintptr_t)&Bs[r1 * LDS_STRIDE + x1 * 8];

  const __bf16* gA = A + (size_t)m0 * K;        // A slab origin (row m0)
  uint32_t ldsA = (uint32_t)(uintptr_t)&As[0];

  v8f acc[4][2] = {};

  for (int k = 0; k < K; k += 32) {
    if (wave == 0) {
      // TDM: 128 rows x 32 bf16 from A[m0..m0+127][k..k+31] into padded LDS.
      tdm_load_slab_bf16(ldsA, gA + k, /*row_elems=*/32, /*rows=*/128,
                         /*ld=*/(uint32_t)K, /*tensor_d0=*/(uint32_t)(K - k),
                         /*tensor_d1=*/(uint32_t)M);
    }
    async_copy_b128(lb0, gb0 + k);
    async_copy_b128(lb1, gb1 + k);
    wait_async_all();
    if (wave == 0) __builtin_amdgcn_s_wait_tensorcnt(0);
    __syncthreads();

    v16bf a[4], b[2];
#pragma unroll
    for (int i = 0; i < 4; ++i) a[i] = load_a16(As, LDS_STRIDE, wm * 64 + i * 16, 0);
#pragma unroll
    for (int j = 0; j < 2; ++j) b[j] = load_b16(Bs, LDS_STRIDE, wn * 32 + j * 16, 0);
#pragma unroll
    for (int i = 0; i < 4; ++i)
#pragma unroll
      for (int j = 0; j < 2; ++j)
        acc[i][j] = wmma_bf16(a[i], b[j], acc[i][j]);

    __syncthreads();
  }

  // Store C-layout tiles: lane holds col = lane&15, rows v + 8*(lane>>4).
  int colb = lane & 15, half = lane >> 4;
#pragma unroll
  for (int j = 0; j < 2; ++j) {
    int cc = n0 + wn * 32 + j * 16 + colb;
    float bv = bias ? bias[cc] : 0.0f;
#pragma unroll
    for (int i = 0; i < 4; ++i) {
      int rbase = m0 + wm * 64 + i * 16 + (half << 3);
#pragma unroll
      for (int v = 0; v < 8; ++v) {
        float r = acc[i][j][v] + bv;
        C[(size_t)(rbase + v) * N + cc] = (OutT)r;
      }
    }
  }
}

// ---------------------------------------------------------------------------
// RoPE + head split. qkv is (B*T, 3C) bf16. Produces:
//   qh, kh : (B,H,T,D) bf16   (q pre-scaled by 1/sqrt(D))
//   vT     : (B,H,D,T) bf16   (transposed so PV B-tiles load contiguously)
// Each thread handles one (b,h,t,d) pair with d in [0,32).
// ---------------------------------------------------------------------------
__global__ void rope_split_kernel(const __bf16* __restrict__ qkv,
                                  const float* __restrict__ cosb,
                                  const float* __restrict__ sinb,
                                  __bf16* __restrict__ qh,
                                  __bf16* __restrict__ kh,
                                  __bf16* __restrict__ vT) {
  int tid = blockIdx.x * blockDim.x + threadIdx.x;   // B*H*T*32 threads
  int d  = tid & 31;
  int t  = (tid >> 5) & (TT - 1);
  int bh = tid >> 16;              // 5 + 11 bits
  int h  = bh & (HH - 1);
  int b  = bh >> 4;

  size_t row = (size_t)(b * TT + t) * N_QKV;
  int cb = h * DD;

  float c0 = cosb[t * DD + d],      s0 = sinb[t * DD + d];
  float c1 = cosb[t * DD + d + 32], s1 = sinb[t * DD + d + 32];

  float qa = (float)qkv[row + cb + d];
  float qb = (float)qkv[row + cb + d + 32];
  float ka = (float)qkv[row + CC + cb + d];
  float kb = (float)qkv[row + CC + cb + d + 32];
  float va = (float)qkv[row + 2 * CC + cb + d];
  float vb = (float)qkv[row + 2 * CC + cb + d + 32];

  const float sc = 0.125f;  // 1/sqrt(64) folded into q
  size_t o = ((size_t)bh * TT + t) * DD;
  qh[o + d]      = (__bf16)((qa * c0 - qb * s0) * sc);
  qh[o + d + 32] = (__bf16)((qb * c1 + qa * s1) * sc);
  kh[o + d]      = (__bf16)(ka * c0 - kb * s0);
  kh[o + d + 32] = (__bf16)(kb * c1 + ka * s1);

  size_t ov = (size_t)bh * DD * TT + t;
  vT[ov + (size_t)d * TT]        = (__bf16)va;
  vT[ov + (size_t)(d + 32) * TT] = (__bf16)vb;
}

// ---------------------------------------------------------------------------
// Flash-style causal attention. One wave handles a 16-query tile of one head,
// streaming 32-key blocks: 4 WMMAs for scores, online softmax (shfl_xor row
// reductions inside the 16-lane C-layout groups), P staged via LDS to convert
// C-layout f32 -> A-layout bf16, then 4 WMMAs accumulate P*V (D=64).
// Output written as bf16 context in (B, T, C) layout.
// ---------------------------------------------------------------------------
__global__ __launch_bounds__(128) void flash_attn_kernel(
    const __bf16* __restrict__ qh, const __bf16* __restrict__ kh,
    const __bf16* __restrict__ vT, __bf16* __restrict__ ctx) {
  __shared__ __attribute__((aligned(16))) __bf16 plds[4][16 * 40];  // 40-elt row stride

  int wave = threadIdx.x >> 5;
  int lane = threadIdx.x & 31;
  int gid = blockIdx.x * 4 + wave;          // 0 .. B*H*(T/16)-1
  int qt = gid & (TT / 16 - 1);
  int bh = gid >> 7;                        // T/16 = 128 -> 7 bits
  int q0 = qt << 4;

  const __bf16* qp = qh + (size_t)bh * TT * DD;
  const __bf16* kp = kh + (size_t)bh * TT * DD;
  const __bf16* vp = vT + (size_t)bh * DD * TT;

  v16bf aq0 = load_a16(qp, DD, q0, 0);
  v16bf aq1 = load_a16(qp, DD, q0, 32);

  float m[8], l[8];
  v8f occ[4] = {};
#pragma unroll
  for (int v = 0; v < 8; ++v) { m[v] = -1e30f; l[v] = 0.0f; }

  int col = lane & 15;
  int half = lane >> 4;
  __bf16* pl = plds[wave];

  for (int kb = 0; kb < q0 + 16; kb += 32) {
    // ---- scores: S(16x32) = Q(16x64) * K^T, as two 16x16 tiles ----
    v8f s0 = {}, s1 = {};
    {
      v16bf bk = load_b16(kp, DD, kb, 0);       s0 = wmma_bf16(aq0, bk, s0);
      bk       = load_b16(kp, DD, kb, 32);      s0 = wmma_bf16(aq1, bk, s0);
      bk       = load_b16(kp, DD, kb + 16, 0);  s1 = wmma_bf16(aq0, bk, s1);
      bk       = load_b16(kp, DD, kb + 16, 32); s1 = wmma_bf16(aq1, bk, s1);
    }

    // ---- causal mask + row max ----
    float e0[8], e1[8], mx[8];
#pragma unroll
    for (int v = 0; v < 8; ++v) {
      int row = q0 + v + (half << 3);
      float a0 = (kb + col      <= row) ? s0[v] : -1e30f;
      float a1 = (kb + 16 + col <= row) ? s1[v] : -1e30f;
      e0[v] = a0; e1[v] = a1;
      mx[v] = fmaxf(a0, a1);
    }
#pragma unroll
    for (int off = 1; off < 16; off <<= 1)
#pragma unroll
      for (int v = 0; v < 8; ++v)
        mx[v] = fmaxf(mx[v], __shfl_xor(mx[v], off, 32));

    // ---- online softmax update; stage P to LDS as bf16 ----
    float rs[8];
#pragma unroll
    for (int v = 0; v < 8; ++v) {
      float mn = fmaxf(m[v], mx[v]);
      float scale = __expf(m[v] - mn);
      m[v] = mn;
      float p0 = __expf(e0[v] - mn);
      float p1 = __expf(e1[v] - mn);
      rs[v] = p0 + p1;
      int row = v + (half << 3);
      pl[row * 40 + col]      = (__bf16)p0;
      pl[row * 40 + 16 + col] = (__bf16)p1;
      l[v] *= scale;
#pragma unroll
      for (int t = 0; t < 4; ++t) occ[t][v] *= scale;
    }
#pragma unroll
    for (int off = 1; off < 16; off <<= 1)
#pragma unroll
      for (int v = 0; v < 8; ++v)
        rs[v] += __shfl_xor(rs[v], off, 32);
#pragma unroll
    for (int v = 0; v < 8; ++v) l[v] += rs[v];

    // intra-wave LDS RAW: drain DS pipe before the transposed reload
    asm volatile("s_wait_dscnt 0" ::: "memory");

    // ---- context accumulate: O(16x64) += P(16x32) * V(32x64) ----
    v16bf pa = load_a16(pl, 40, 0, 0);
#pragma unroll
    for (int t = 0; t < 4; ++t) {
      v16bf bv = load_b16(vp, TT, t * 16, kb);
      occ[t] = wmma_bf16(pa, bv, occ[t]);
    }
    asm volatile("s_wait_dscnt 0" ::: "memory");
  }

  // ---- normalize and write context in (B, T, C) ----
  int h = bh & (HH - 1), b = bh >> 4;
#pragma unroll
  for (int v = 0; v < 8; ++v) {
    float inv = 1.0f / l[v];
    int row = q0 + v + (half << 3);
    size_t base = (size_t)(b * TT + row) * CC + h * DD;
#pragma unroll
    for (int t = 0; t < 4; ++t)
      ctx[base + t * 16 + col] = (__bf16)(occ[t][v] * inv);
  }
}

// ---------------------------------------------------------------------------
// Host launch
// ---------------------------------------------------------------------------
extern "C" void kernel_launch(void* const* d_in, const int* in_sizes, int n_in,
                              void* d_out, int out_size, void* d_ws, size_t ws_size,
                              hipStream_t stream) {
  const float* x    = (const float*)d_in[0];
  const float* cosb = (const float*)d_in[1];
  const float* sinb = (const float*)d_in[2];
  // d_in[3] = attn_mask: unused, causal mask computed analytically
  const float* wqkv = (const float*)d_in[4];
  const float* bqkv = (const float*)d_in[5];
  const float* wout = (const float*)d_in[6];
  const float* bout = (const float*)d_in[7];
  float* out = (float*)d_out;

  char* ws = (char*)d_ws;
  size_t off = 0;
  auto take = [&](size_t elems) -> __bf16* {
    __bf16* p = (__bf16*)(ws + off);
    off = (off + elems * sizeof(__bf16) + 255) & ~(size_t)255;
    return p;
  };

  __bf16* xb    = take((size_t)MM * CC);          // x in bf16
  __bf16* wqkvb = take((size_t)N_QKV * CC);       // qkv weight bf16 (N-major)
  __bf16* woutb = take((size_t)CC * CC);          // out weight bf16 (N-major)
  __bf16* qkvb  = take((size_t)MM * N_QKV);       // qkv activations bf16
  __bf16* qhb   = take((size_t)BB * HH * TT * DD);
  __bf16* khb   = take((size_t)BB * HH * TT * DD);
  __bf16* vTb   = take((size_t)BB * HH * DD * TT);
  __bf16* ctxb  = take((size_t)MM * CC);          // attention context bf16
  (void)ws_size; (void)in_sizes; (void)n_in; (void)out_size;

  // 1) precision conversion
  {
    int n = MM * CC;
    f32_to_bf16_kernel<<<(n + 255) / 256, 256, 0, stream>>>(x, xb, n);
    n = N_QKV * CC;
    f32_to_bf16_kernel<<<(n + 255) / 256, 256, 0, stream>>>(wqkv, wqkvb, n);
    n = CC * CC;
    f32_to_bf16_kernel<<<(n + 255) / 256, 256, 0, stream>>>(wout, woutb, n);
  }

  // 2) QKV projection: (4096 x 3072) = x(4096x1024) * Wqkv^T + bias
  wmma_gemm_async<__bf16><<<dim3(N_QKV / 128, MM / 128), 256, 0, stream>>>(
      xb, wqkvb, bqkv, qkvb, MM, N_QKV, CC);

  // 3) RoPE + head split (+ V transpose, + q scaling)
  {
    int n = BB * HH * TT * 32;
    rope_split_kernel<<<n / 256, 256, 0, stream>>>(qkvb, cosb, sinb, qhb, khb, vTb);
  }

  // 4) causal flash attention -> context (B,T,C) bf16
  flash_attn_kernel<<<(BB * HH * (TT / 16)) / 4, 128, 0, stream>>>(qhb, khb, vTb, ctxb);

  // 5) output projection: out(4096x1024) = ctx * Wout^T + bias (f32 store)
  wmma_gemm_async<float><<<dim3(CC / 128, MM / 128), 256, 0, stream>>>(
      ctxb, woutb, bout, out, MM, CC, CC);
}
